// UNet1D_33440615367250
// MI455X (gfx1250) — matrix operand order repaired and checked by reading, not compiled, not run on hardware
//
#include <hip/hip_runtime.h>
#include <math.h>

// ---------------------------------------------------------------------------
// CDNA5 (gfx1250) UNet1D forward. All dense GEMMs go through
// v_wmma_f32_16x16x32_f16, wave32, 2x2 register-blocked 32x32 C tile per wave
// (4 WMMAs per K-chunk), f32 accumulate. All addressing is 32-bit.
// ---------------------------------------------------------------------------

typedef __attribute__((ext_vector_type(16))) _Float16 v16h;
typedef __attribute__((ext_vector_type(8)))  float   v8f;

union HFrag { v16h v; _Float16 h[16]; };

// ----------------------------- generic WMMA GEMM ---------------------------
// C[m,n] = sum_k A[m,k]*B[k,n]  (+bias1[n]) (+bias2[n]) (+D[m*d_mul+n]), act.
// A addressing: off = (m/a_div)*a_mul1 + (m%a_div)*a_mul2 + k*a_k
// B addressing: off = k*b_k + n*b_n
// C addressing: off = (m/c_div)*c_mul1 + (m%c_div)*c_mul2 + n*c_n
struct GemmParams {
  const float* A; const float* B; float* C;
  const float* bias1; const float* bias2; const float* D;
  int a_div, a_mul1, a_mul2, a_k;
  int b_k, b_n;
  int c_div, c_mul1, c_mul2, c_n;
  int d_mul;
  int M, N, K, act;   // act: 0=none 1=relu 2=gelu(exact)
};

__device__ __forceinline__ int a_kidx(int e, int g) {
  // 16-bit A fragment (16x32): lanes 0-15: K=e; lanes 16-31: +16
  return e + (g << 4);
}

template <bool AV>
__device__ __forceinline__ v16h load_a_frag(const float* A, unsigned aro,
                                            int kb, int ak, int g) {
  HFrag f;
  if (AV) {
    const float* base = A + aro + kb + (g << 4);
    float4 r0 = *(const float4*)(base);
    float4 r1 = *(const float4*)(base + 4);
    float4 r2 = *(const float4*)(base + 8);
    float4 r3 = *(const float4*)(base + 12);
    f.h[0]=(_Float16)r0.x;  f.h[1]=(_Float16)r0.y;
    f.h[2]=(_Float16)r0.z;  f.h[3]=(_Float16)r0.w;
    f.h[4]=(_Float16)r1.x;  f.h[5]=(_Float16)r1.y;
    f.h[6]=(_Float16)r1.z;  f.h[7]=(_Float16)r1.w;
    f.h[8]=(_Float16)r2.x;  f.h[9]=(_Float16)r2.y;
    f.h[10]=(_Float16)r2.z; f.h[11]=(_Float16)r2.w;
    f.h[12]=(_Float16)r3.x; f.h[13]=(_Float16)r3.y;
    f.h[14]=(_Float16)r3.z; f.h[15]=(_Float16)r3.w;
  } else {
#pragma unroll
    for (int e = 0; e < 16; ++e)
      f.h[e] = (_Float16)A[aro + (unsigned)((kb + a_kidx(e, g)) * ak)];
  }
  return f.v;
}

template <bool BV>
__device__ __forceinline__ v16h load_b_frag(const float* B, unsigned bco,
                                            int kb, int bk, int g) {
  HFrag f;
  if (BV) {
    const float* base = B + bco + kb + (g << 4);
    float4 r0 = *(const float4*)(base);
    float4 r1 = *(const float4*)(base + 4);
    float4 r2 = *(const float4*)(base + 8);
    float4 r3 = *(const float4*)(base + 12);
    f.h[0]=(_Float16)r0.x;  f.h[1]=(_Float16)r0.y;
    f.h[2]=(_Float16)r0.z;  f.h[3]=(_Float16)r0.w;
    f.h[4]=(_Float16)r1.x;  f.h[5]=(_Float16)r1.y;
    f.h[6]=(_Float16)r1.z;  f.h[7]=(_Float16)r1.w;
    f.h[8]=(_Float16)r2.x;  f.h[9]=(_Float16)r2.y;
    f.h[10]=(_Float16)r2.z; f.h[11]=(_Float16)r2.w;
    f.h[12]=(_Float16)r3.x; f.h[13]=(_Float16)r3.y;
    f.h[14]=(_Float16)r3.z; f.h[15]=(_Float16)r3.w;
  } else {
#pragma unroll
    for (int e = 0; e < 16; ++e)
      f.h[e] = (_Float16)B[bco + (unsigned)((kb + e + (g << 4)) * bk)];
  }
  return f.v;
}

// Tail fragments: clamped addresses + selects, no divergent branches.
__device__ __forceinline__ v16h load_a_tail(const float* A, unsigned aro,
                                            int kb, int ak, int g, int K) {
  HFrag f; int Km1 = K - 1;
#pragma unroll
  for (int e = 0; e < 16; ++e) {
    int k = kb + a_kidx(e, g);
    int kc = k < Km1 ? k : Km1;
    float fv = A[aro + (unsigned)(kc * ak)];
    f.h[e] = (_Float16)(k < K ? fv : 0.f);
  }
  return f.v;
}
__device__ __forceinline__ v16h load_b_tail(const float* B, unsigned bco,
                                            int kb, int bk, int g, int K) {
  HFrag f; int Km1 = K - 1;
#pragma unroll
  for (int e = 0; e < 16; ++e) {
    int k = kb + e + (g << 4);
    int kc = k < Km1 ? k : Km1;
    float fv = B[bco + (unsigned)(kc * bk)];
    f.h[e] = (_Float16)(k < K ? fv : 0.f);
  }
  return f.v;
}

// AV: A fragment loads vectorizable (a_k==1, 16B-aligned rows)
// BV: B fragment loads vectorizable (b_k==1, 16B-aligned columns)
template <bool AV, bool BV>
__global__ __launch_bounds__(256) void gemm_wmma_kernel(GemmParams p) {
  const int wave = threadIdx.x >> 5;
  const int lane = threadIdx.x & 31;
  const int g    = lane >> 4;       // lane group (0: lanes 0-15, 1: 16-31)
  const int l4   = lane & 15;

  const int m0 = blockIdx.x * 32;
  const int n0 = (blockIdx.y * 8 + wave) * 32;
  if (n0 >= p.N) return;            // wave-uniform exit (EXEC stays all-ones)

  // Per-lane A rows for the two m-subtiles (clamped; stores guarded later).
  int mA0 = m0 + l4;      if (mA0 >= p.M) mA0 = p.M - 1;
  int mA1 = m0 + 16 + l4; if (mA1 >= p.M) mA1 = p.M - 1;
  const unsigned aro0 = (unsigned)(mA0 / p.a_div) * p.a_mul1 +
                        (unsigned)(mA0 % p.a_div) * p.a_mul2;
  const unsigned aro1 = (unsigned)(mA1 / p.a_div) * p.a_mul1 +
                        (unsigned)(mA1 % p.a_div) * p.a_mul2;
  // Per-lane B columns for the two n-subtiles (clamped).
  int nB0 = n0 + l4;      if (nB0 >= p.N) nB0 = p.N - 1;
  int nB1 = n0 + 16 + l4; if (nB1 >= p.N) nB1 = p.N - 1;
  const unsigned bco0 = (unsigned)nB0 * p.b_n;
  const unsigned bco1 = (unsigned)nB1 * p.b_n;

  v8f acc00 = {0.f,0.f,0.f,0.f,0.f,0.f,0.f,0.f};
  v8f acc01 = acc00, acc10 = acc00, acc11 = acc00;

  const int kmain = p.K & ~31;      // full 32-chunks: no guards at all
  for (int kb = 0; kb < kmain; kb += 32) {
    if (kb + 32 < p.K) {
      __builtin_prefetch((const void*)(p.A + aro0 + (unsigned)((kb + 32) * p.a_k)), 0, 1);
      __builtin_prefetch((const void*)(p.B + bco0 + (unsigned)((kb + 32) * p.b_k)), 0, 1);
    }
    v16h a0 = load_a_frag<AV>(p.A, aro0, kb, p.a_k, g);
    v16h a1 = load_a_frag<AV>(p.A, aro1, kb, p.a_k, g);
    v16h b0 = load_b_frag<BV>(p.B, bco0, kb, p.b_k, g);
    v16h b1 = load_b_frag<BV>(p.B, bco1, kb, p.b_k, g);
    acc00 = __builtin_amdgcn_wmma_f32_16x16x32_f16(false, a0, false, b0, (short)0, acc00, false, false);
    acc01 = __builtin_amdgcn_wmma_f32_16x16x32_f16(false, a0, false, b1, (short)0, acc01, false, false);
    acc10 = __builtin_amdgcn_wmma_f32_16x16x32_f16(false, a1, false, b0, (short)0, acc10, false, false);
    acc11 = __builtin_amdgcn_wmma_f32_16x16x32_f16(false, a1, false, b1, (short)0, acc11, false, false);
  }

  if (kmain < p.K) {                // K tail (K%32 != 0)
    const int kb = kmain;
    v16h a0 = load_a_tail(p.A, aro0, kb, p.a_k, g, p.K);
    v16h a1 = load_a_tail(p.A, aro1, kb, p.a_k, g, p.K);
    v16h b0 = load_b_tail(p.B, bco0, kb, p.b_k, g, p.K);
    v16h b1 = load_b_tail(p.B, bco1, kb, p.b_k, g, p.K);
    acc00 = __builtin_amdgcn_wmma_f32_16x16x32_f16(false, a0, false, b0, (short)0, acc00, false, false);
    acc01 = __builtin_amdgcn_wmma_f32_16x16x32_f16(false, a0, false, b1, (short)0, acc01, false, false);
    acc10 = __builtin_amdgcn_wmma_f32_16x16x32_f16(false, a1, false, b0, (short)0, acc10, false, false);
    acc11 = __builtin_amdgcn_wmma_f32_16x16x32_f16(false, a1, false, b1, (short)0, acc11, false, false);
  }

  // Epilogue. C/D 16x16 f32 layout: VGPR r -> M = r + 8*g, lanes -> N.
  v8f accs[2][2] = {{acc00, acc01}, {acc10, acc11}};
#pragma unroll
  for (int ti = 0; ti < 2; ++ti) {
#pragma unroll
    for (int tj = 0; tj < 2; ++tj) {
      int nn = n0 + tj * 16 + l4;
#pragma unroll
      for (int r = 0; r < 8; ++r) {
        int m = m0 + ti * 16 + r + (g << 3);
        if (m < p.M && nn < p.N) {
          float v = accs[ti][tj][r];
          if (p.bias1) v += p.bias1[nn];
          if (p.bias2) v += p.bias2[nn];
          if (p.D)     v += p.D[(unsigned)(m * p.d_mul) + (unsigned)nn];
          if (p.act == 1)      v = fmaxf(v, 0.f);
          else if (p.act == 2) v = 0.5f * v * (1.f + erff(v * 0.70710678118654752f));
          unsigned co = (unsigned)(m / p.c_div) * p.c_mul1 +
                        (unsigned)(m % p.c_div) * p.c_mul2 +
                        (unsigned)nn * p.c_n;
          p.C[co] = v;
        }
      }
    }
  }
}

// ------------------------------- LSTM cell ---------------------------------
__global__ void lstm_cell_kernel(const float* gates, float* cst, float* hst,
                                 float* out, int Bn, int H, int L, int t,
                                 int dirOff) {
  int idx = blockIdx.x * blockDim.x + threadIdx.x;
  if (idx >= Bn * H) return;
  int b = idx / H, h = idx % H;
  const float* z = gates + (size_t)b * 4 * H;
  float i = 1.f / (1.f + expf(-z[h]));
  float f = 1.f / (1.f + expf(-z[H + h]));
  float g = tanhf(z[2 * H + h]);
  float o = 1.f / (1.f + expf(-z[3 * H + h]));
  float c = f * cst[idx] + i * g;
  cst[idx] = c;
  float hv = o * tanhf(c);
  hst[idx] = hv;
  out[(size_t)b * (2 * H) * L + (size_t)(dirOff + h) * L + t] = hv;
}

// ------------------------------- GAT kernels -------------------------------
__device__ __forceinline__ unsigned mkey_enc(float f) {
  unsigned u = __float_as_uint(f);
  return (u & 0x80000000u) ? ~u : (u | 0x80000000u);
}
__device__ __forceinline__ float mkey_dec(unsigned k) {
  unsigned u = (k & 0x80000000u) ? (k & 0x7fffffffu) : ~k;
  return __uint_as_float(u);
}
__device__ __forceinline__ void edge_ends(const int* ei, int E, int e,
                                          int& s, int& d) {
  if (e < E) { s = ei[e]; d = ei[E + e]; }
  else       { s = d = e - E; }            // self loops
}

__global__ void gat_att_kernel(const float* h, const float* asrcW,
                               const float* adstW, float* asrc, float* adst,
                               int N, int outc) {
  int idx = blockIdx.x * blockDim.x + threadIdx.x;
  if (idx >= N * 3) return;
  int n = idx / 3, hd = idx % 3;
  const float* hp = h + ((size_t)n * 3 + hd) * outc;
  const float* ws = asrcW + (size_t)hd * outc;
  const float* wd = adstW + (size_t)hd * outc;
  float s = 0.f, d = 0.f;
  for (int c = 0; c < outc; ++c) { float hv = hp[c]; s += hv * ws[c]; d += hv * wd[c]; }
  asrc[idx] = s; adst[idx] = d;
}

__global__ void gat_edge_max_kernel(const int* ei, int E, int N,
                                    const float* asrc, const float* adst,
                                    unsigned* mkey) {
  int idx = blockIdx.x * blockDim.x + threadIdx.x;
  int Etot = E + N;
  if (idx >= Etot * 3) return;
  int e = idx / 3, hd = idx % 3, s, d;
  edge_ends(ei, E, e, s, d);
  float v = asrc[s * 3 + hd] + adst[d * 3 + hd];
  v = (v > 0.f) ? v : 0.2f * v;
  atomicMax(&mkey[d * 3 + hd], mkey_enc(v));
}

__global__ void gat_edge_sum_kernel(const int* ei, int E, int N,
                                    const float* asrc, const float* adst,
                                    const unsigned* mkey, float* denom) {
  int idx = blockIdx.x * blockDim.x + threadIdx.x;
  int Etot = E + N;
  if (idx >= Etot * 3) return;
  int e = idx / 3, hd = idx % 3, s, d;
  edge_ends(ei, E, e, s, d);
  float v = asrc[s * 3 + hd] + adst[d * 3 + hd];
  v = (v > 0.f) ? v : 0.2f * v;
  atomicAdd(&denom[d * 3 + hd], expf(v - mkey_dec(mkey[d * 3 + hd])));
}

__global__ void gat_edge_aggr_kernel(const int* ei, int E, int N,
                                     const float* asrc, const float* adst,
                                     const unsigned* mkey, const float* denom,
                                     const float* h, float* acc, int outc) {
  int e = blockIdx.x / 3, hd = blockIdx.x % 3, s, d;
  edge_ends(ei, E, e, s, d);
  float v = asrc[s * 3 + hd] + adst[d * 3 + hd];
  v = (v > 0.f) ? v : 0.2f * v;
  float ex = expf(v - mkey_dec(mkey[d * 3 + hd]));
  float alpha = ex / (denom[d * 3 + hd] + 1e-16f);
  const float* hs = h + ((size_t)s * 3 + hd) * outc;
  float* ad = acc + ((size_t)d * 3 + hd) * outc;
  for (int c = threadIdx.x; c < outc; c += blockDim.x)
    atomicAdd(&ad[c], alpha * hs[c]);
}

__global__ void gat_finalize_kernel(const float* acc, const float* bias,
                                    const float* g, const float* b,
                                    const float* rm, const float* rv,
                                    float* out, int N, int outc) {
  size_t idx = (size_t)blockIdx.x * blockDim.x + threadIdx.x;
  if (idx >= (size_t)N * outc) return;
  int n = (int)(idx / outc), c = (int)(idx % outc);
  const float* a = acc + (size_t)n * 3 * outc;
  float v = (a[c] + a[outc + c] + a[2 * outc + c]) * (1.f / 3.f) + bias[c];
  v = (v - rm[c]) * rsqrtf(rv[c] + 1e-5f) * g[c] + b[c];
  out[idx] = fmaxf(v, 0.f);
}

// ------------------------------- convolutions ------------------------------
__global__ void conv1d_s2_kernel(const float* x, const float* w, const float* bias,
                                 float* y, int Bn, int Cin, int Lin, int Cout) {
  int Lout = Lin / 2;
  size_t idx = (size_t)blockIdx.x * blockDim.x + threadIdx.x;
  size_t tot = (size_t)Bn * Cout * Lout;
  if (idx >= tot) return;
  int t = (int)(idx % Lout); int r = (int)(idx / Lout);
  int o = r % Cout, b = r / Cout;
  int tin = 2 * t - 1;
  float acc = bias[o];
  for (int c = 0; c < Cin; ++c) {
    const float* xr = x + ((size_t)b * Cin + c) * Lin;
    const float* wr = w + ((size_t)o * Cin + c) * 3;
    if (tin >= 0)       acc += xr[tin]     * wr[0];
                        acc += xr[tin + 1] * wr[1];
    if (tin + 2 < Lin)  acc += xr[tin + 2] * wr[2];
  }
  y[idx] = acc;
}

__global__ void convt_k2s2_kernel(const float* x, const float* w, const float* bias,
                                  float* y, int Bn, int Cin, int Lin, int Cout) {
  int Lout = 2 * Lin;
  size_t idx = (size_t)blockIdx.x * blockDim.x + threadIdx.x;
  size_t tot = (size_t)Bn * Cout * Lout;
  if (idx >= tot) return;
  int l = (int)(idx % Lout); int r = (int)(idx / Lout);
  int o = r % Cout, b = r / Cout;
  int t = l >> 1, k = l & 1;
  float acc = bias[o];
  for (int i = 0; i < Cin; ++i)
    acc += x[((size_t)b * Cin + i) * Lin + t] * w[((size_t)i * Cout + o) * 2 + k];
  y[idx] = acc;
}

// -------------------------- cross attention pieces -------------------------
__global__ void attn_scores_kernel(const float* q, const float* k, float* sc,
                                   int Bn) {
  int idx = blockIdx.x * blockDim.x + threadIdx.x;
  if (idx >= Bn * 4 * 12 * 12) return;
  int ki = idx % 12; int t = idx / 12;
  int qi = t % 12; t /= 12;
  int h = t % 4, b = t / 4;
  const float* qp = q + ((size_t)b * 12 + qi) * 1024 + h * 256;
  const float* kp = k + ((size_t)b * 12 + ki) * 1024 + h * 256;
  float acc = 0.f;
  for (int d = 0; d < 256; ++d) acc += qp[d] * kp[d];
  sc[idx] = acc * 0.0625f;   // 1/sqrt(256)
}

__global__ void attn_softmax_kernel(float* sc, int Bn) {
  int row = blockIdx.x * blockDim.x + threadIdx.x;
  if (row >= Bn * 4 * 12) return;
  float* p = sc + (size_t)row * 12;
  float mx = p[0];
  for (int i = 1; i < 12; ++i) mx = fmaxf(mx, p[i]);
  float s = 0.f;
  for (int i = 0; i < 12; ++i) { float e = expf(p[i] - mx); p[i] = e; s += e; }
  float inv = 1.f / s;
  for (int i = 0; i < 12; ++i) p[i] *= inv;
}

__global__ void attn_av_kernel(const float* sc, const float* v, float* o,
                               int Bn) {
  size_t idx = (size_t)blockIdx.x * blockDim.x + threadIdx.x;
  if (idx >= (size_t)Bn * 12 * 1024) return;
  int n = (int)(idx % 1024); int t = (int)(idx / 1024);
  int qi = t % 12, b = t / 12, h = n >> 8;
  const float* srow = sc + (((size_t)b * 4 + h) * 12 + qi) * 12;
  float acc = 0.f;
  for (int ki = 0; ki < 12; ++ki)
    acc += srow[ki] * v[((size_t)b * 12 + ki) * 1024 + n];
  o[idx] = acc;
}

__global__ void layernorm_kernel(const float* x, const float* g, const float* b,
                                 float* y, int E) {
  int row = blockIdx.x;
  const float* xp = x + (size_t)row * E;
  __shared__ float red[256];
  float s = 0.f;
  for (int i = threadIdx.x; i < E; i += 256) s += xp[i];
  red[threadIdx.x] = s; __syncthreads();
  for (int o = 128; o > 0; o >>= 1) {
    if (threadIdx.x < o) red[threadIdx.x] += red[threadIdx.x + o];
    __syncthreads();
  }
  float mu = red[0] / E; __syncthreads();
  float v = 0.f;
  for (int i = threadIdx.x; i < E; i += 256) { float d = xp[i] - mu; v += d * d; }
  red[threadIdx.x] = v; __syncthreads();
  for (int o = 128; o > 0; o >>= 1) {
    if (threadIdx.x < o) red[threadIdx.x] += red[threadIdx.x + o];
    __syncthreads();
  }
  float inv = rsqrtf(red[0] / E + 1e-5f);
  for (int i = threadIdx.x; i < E; i += 256)
    y[(size_t)row * E + i] = (xp[i] - mu) * inv * g[i] + b[i];
}

__global__ void transpose_b_kernel(const float* x, float* y, int Bn, int d1,
                                   int d2) {           // [Bn,d1,d2]->[Bn,d2,d1]
  size_t idx = (size_t)blockIdx.x * blockDim.x + threadIdx.x;
  size_t tot = (size_t)Bn * d1 * d2;
  if (idx >= tot) return;
  int j = (int)(idx % d2); int r = (int)(idx / d2);
  int i = r % d1, b = r / d1;
  y[((size_t)b * d2 + j) * d1 + i] = x[idx];
}

// ------------------------------- host helpers ------------------------------
static void gemm(hipStream_t s,
                 const float* A, int adiv, int am1, int am2, int ak,
                 const float* Bm, int bk, int bn,
                 float* C, int cdiv, int cm1, int cm2, int cn,
                 int M, int N, int K,
                 const float* b1 = nullptr, const float* b2 = nullptr,
                 const float* D = nullptr, int dmul = 0, int act = 0) {
  GemmParams p;
  p.A = A; p.B = Bm; p.C = C; p.bias1 = b1; p.bias2 = b2; p.D = D;
  p.a_div = adiv; p.a_mul1 = am1; p.a_mul2 = am2; p.a_k = ak;
  p.b_k = bk; p.b_n = bn;
  p.c_div = cdiv; p.c_mul1 = cm1; p.c_mul2 = cm2; p.c_n = cn;
  p.d_mul = dmul; p.M = M; p.N = N; p.K = K; p.act = act;
  dim3 grid((M + 31) / 32, (N + 255) / 256);
  bool av = (ak == 1) && (am1 % 4 == 0) && (am2 % 4 == 0);
  bool bv = (bk == 1) && (bn % 4 == 0);
  if (av && bv)      gemm_wmma_kernel<true,  true ><<<grid, 256, 0, s>>>(p);
  else if (av)       gemm_wmma_kernel<true,  false><<<grid, 256, 0, s>>>(p);
  else if (bv)       gemm_wmma_kernel<false, true ><<<grid, 256, 0, s>>>(p);
  else               gemm_wmma_kernel<false, false><<<grid, 256, 0, s>>>(p);
}

struct LW { const float *wih_f,*whh_f,*bih_f,*bhh_f,*wih_b,*whh_b,*bih_b,*bhh_b; };
struct GW { const float *W,*as,*ad,*bias,*g,*b,*rm,*rv; };

static void run_bilstm(hipStream_t s, const float* x, int Bn, int din, int L,
                       int H, const LW& w, float* out, float* pre, float* gates,
                       float* hst, float* cst) {
  for (int dir = 0; dir < 2; ++dir) {
    const float* wih = dir ? w.wih_b : w.wih_f;
    const float* whh = dir ? w.whh_b : w.whh_f;
    const float* bih = dir ? w.bih_b : w.bih_f;
    const float* bhh = dir ? w.bhh_b : w.bhh_f;
    // pre[b,t,:] = x[b,:,t] @ Wih^T  for all t (one big WMMA GEMM)
    gemm(s, x, L, din * L, 1, L, wih, 1, din,
         pre, 1, 4 * H, 0, 1, Bn * L, 4 * H, din);
    (void)hipMemsetAsync(hst, 0, (size_t)Bn * H * sizeof(float), s);
    (void)hipMemsetAsync(cst, 0, (size_t)Bn * H * sizeof(float), s);
    for (int step = 0; step < L; ++step) {
      int t = dir ? (L - 1 - step) : step;
      gemm(s, hst, 1, H, 0, 1, whh, 1, H,
           gates, 1, 4 * H, 0, 1, Bn, 4 * H, H,
           bih, bhh, pre + (size_t)t * 4 * H, L * 4 * H);
      int tot = Bn * H;
      lstm_cell_kernel<<<(tot + 255) / 256, 256, 0, s>>>(
          gates, cst, hst, out, Bn, H, L, t, dir ? H : 0);
    }
  }
}

static void run_gat(hipStream_t s, const float* x, int N, int cin, int outc,
                    const GW& w, const int* edge, int E, float* out,
                    float* hbuf, float* accbuf, float* asrc, float* adst,
                    unsigned* mkey, float* denom) {
  gemm(s, x, 1, cin, 0, 1, w.W, 3 * outc, 1,
       hbuf, 1, 3 * outc, 0, 1, N, 3 * outc, cin);
  gat_att_kernel<<<(N * 3 + 255) / 256, 256, 0, s>>>(hbuf, w.as, w.ad,
                                                     asrc, adst, N, outc);
  (void)hipMemsetAsync(mkey, 0, (size_t)N * 3 * sizeof(unsigned), s);
  (void)hipMemsetAsync(denom, 0, (size_t)N * 3 * sizeof(float), s);
  (void)hipMemsetAsync(accbuf, 0, (size_t)N * 3 * outc * sizeof(float), s);
  int Etot = E + N;
  gat_edge_max_kernel<<<(Etot * 3 + 255) / 256, 256, 0, s>>>(edge, E, N, asrc,
                                                             adst, mkey);
  gat_edge_sum_kernel<<<(Etot * 3 + 255) / 256, 256, 0, s>>>(edge, E, N, asrc,
                                                             adst, mkey, denom);
  gat_edge_aggr_kernel<<<Etot * 3, 64, 0, s>>>(edge, E, N, asrc, adst, mkey,
                                               denom, hbuf, accbuf, outc);
  size_t tot = (size_t)N * outc;
  gat_finalize_kernel<<<(unsigned)((tot + 255) / 256), 256, 0, s>>>(
      accbuf, w.bias, w.g, w.b, w.rm, w.rv, out, N, outc);
}

// ------------------------------- entry point -------------------------------
extern "C" void kernel_launch(void* const* d_in, const int* in_sizes, int n_in,
                              void* d_out, int out_size, void* d_ws,
                              size_t ws_size, hipStream_t stream) {
  (void)n_in; (void)out_size; (void)ws_size;
  int pi = 0;
  const float* X      = (const float*)d_in[pi++];        // [B,90,48]
  const float* CTXIN  = (const float*)d_in[pi++];        // [B,512]
  pi++;                                                  // time (dead code)
  const int*   EDGE   = (const int*)d_in[pi++];          // [2,E]
  const int Bn = in_sizes[0] / (90 * 48);
  const int E  = in_sizes[3] / 2;

  auto NXT = [&]() { return (const float*)d_in[pi++]; };
  auto getL = [&]() { LW w; w.wih_f=NXT(); w.whh_f=NXT(); w.bih_f=NXT();
                      w.bhh_f=NXT(); w.wih_b=NXT(); w.whh_b=NXT();
                      w.bih_b=NXT(); w.bhh_b=NXT(); return w; };
  auto getG = [&]() { GW w; w.W=NXT(); w.as=NXT(); w.ad=NXT(); w.bias=NXT();
                      w.g=NXT(); w.b=NXT(); w.rm=NXT(); w.rv=NXT(); return w; };

  const float* ic_w = NXT(); const float* ic_b = NXT();
  LW l1 = getL(); GW g1 = getG();
  const float* d1_w = NXT(); const float* d1_b = NXT();
  LW l2 = getL(); GW g2 = getG();
  const float* d2_w = NXT(); const float* d2_b = NXT();
  LW l3 = getL(); GW g3 = getG();
  const float *lp_w=NXT(), *lp_b=NXT(), *cp_w=NXT(), *cp_b=NXT(),
              *in_w=NXT(), *in_b=NXT(), *out_w=NXT(), *out_b=NXT(),
              *ln_g=NXT(), *ln_b=NXT(), *ff1_w=NXT(), *ff1_b=NXT(),
              *ff2_w=NXT(), *ff2_b=NXT();
  GW g4 = getG(); LW l4 = getL();
  const float* u1_w = NXT(); const float* u1_b = NXT();
  GW g5 = getG(); LW l5 = getL();
  const float* u2_w = NXT(); const float* u2_b = NXT();
  GW g6 = getG(); LW l6 = getL();
  const float* fin_w = NXT(); const float* fin_b = NXT();

  // ---------------- workspace layout (floats) ----------------
  float* WS = (float*)d_ws;
  const size_t SZ_ACT   = (size_t)Bn * 2048 * 12;   // 6.29M @ B=256
  const size_t SZ_PRE   = (size_t)Bn * 24 * 2048;   // 12.58M
  const size_t SZ_GATES = (size_t)Bn * 4096;
  const size_t SZ_STATE = (size_t)Bn * 1024;
  const size_t SZ_GATH  = (size_t)Bn * 12 * 3 * 2048;  // 18.87M
  const size_t SZ_SMALL = (size_t)Bn * 48 * 3;
  float* ACT0  = WS;
  float* ACT1  = ACT0 + SZ_ACT;
  float* PRE   = ACT1 + SZ_ACT;
  float* GATES = PRE + SZ_PRE;
  float* HST   = GATES + SZ_GATES;
  float* CST   = HST + SZ_STATE;
  float* GATH  = CST + SZ_STATE;
  float* GACC  = GATH + SZ_GATH;
  float* ASRC  = GACC + SZ_GATH;
  float* ADST  = ASRC + SZ_SMALL;
  float* DEN   = ADST + SZ_SMALL;
  unsigned* MKEY = (unsigned*)(DEN + SZ_SMALL);
  float* SCORES  = (float*)(MKEY + SZ_SMALL);       // Bn*4*12*12
  // CA aliases (GAT scratch reused; no overlap within CA stage)
  const size_t T = (size_t)Bn * 12 * 1024;
  float* LAT = GATH;        float* CTX = GATH + T;   float* Q    = GATH + 2*T;
  float* K2  = GATH + 3*T;  float* V   = GATH + 4*T; float* OATT = GATH + 5*T;
  float* SBUF = GACC;       float* LATN = GACC + T;  float* FFH  = GACC + 2*T;
  float* CAOUT = PRE;

  // 1) input conv: [B,90,48] -> [B,128,48]
  gemm(stream, X, 48, 90*48, 1, 48, ic_w, 1, 90,
       ACT0, 48, 128*48, 1, 48, Bn*48, 128, 90, ic_b);
  // 2) bilstm1 -> [B,256,48]
  run_bilstm(stream, ACT0, Bn, 128, 48, 128, l1, ACT1, PRE, GATES, HST, CST);
  // 3) gcn1 (N=B*48, c=256, out=256) -> ACT0
  run_gat(stream, ACT1, Bn*48, 256, 256, g1, EDGE, E, ACT0,
          GATH, GACC, ASRC, ADST, MKEY, DEN);
  // 4) down1 -> [B,512,24]
  { size_t tot = (size_t)Bn*512*24;
    conv1d_s2_kernel<<<(unsigned)((tot+255)/256),256,0,stream>>>(
        ACT0, d1_w, d1_b, ACT1, Bn, 256, 48, 512); }
  // 5) bilstm2 -> [B,1024,24]
  run_bilstm(stream, ACT1, Bn, 512, 24, 512, l2, ACT0, PRE, GATES, HST, CST);
  // 6) gcn2 (N=B*24, c=1024, out=512) -> ACT1 = [B,512,24]
  run_gat(stream, ACT0, Bn*24, 1024, 512, g2, EDGE, E, ACT1,
          GATH, GACC, ASRC, ADST, MKEY, DEN);
  // 7) down2 -> [B,1024,12]
  { size_t tot = (size_t)Bn*1024*12;
    conv1d_s2_kernel<<<(unsigned)((tot+255)/256),256,0,stream>>>(
        ACT1, d2_w, d2_b, ACT0, Bn, 512, 24, 1024); }
  // 8) bilstm3 -> [B,2048,12]
  run_bilstm(stream, ACT0, Bn, 1024, 12, 1024, l3, ACT1, PRE, GATES, HST, CST);
  // 9) gcn3 (N=B*12, c=2048, out=2048) -> ACT0 = [B,2048,12]
  run_gat(stream, ACT1, Bn*12, 2048, 2048, g3, EDGE, E, ACT0,
          GATH, GACC, ASRC, ADST, MKEY, DEN);
  // 10) cross attention on ACT0 viewed as latent [B,12,2048]
  gemm(stream, ACT0, 12, 2048*12, 1, 12, lp_w, 1, 2048,
       LAT, 1, 1024, 0, 1, Bn*12, 1024, 2048, lp_b);
  gemm(stream, CTXIN, 1, 512, 0, 1, cp_w, 1, 512,
       CTX, 1, 12288, 0, 1, Bn, 12288, 512, cp_b);
  gemm(stream, LAT, 1, 1024, 0, 1, in_w, 1, 1024,
       Q, 1, 1024, 0, 1, Bn*12, 1024, 1024, in_b);
  gemm(stream, CTX, 1, 1024, 0, 1, in_w + 1024L*1024, 1, 1024,
       K2, 1, 1024, 0, 1, Bn*12, 1024, 1024, in_b + 1024);
  gemm(stream, CTX, 1, 1024, 0, 1, in_w + 2L*1024*1024, 1, 1024,
       V, 1, 1024, 0, 1, Bn*12, 1024, 1024, in_b + 2048);
  { int tot = Bn*4*12*12;
    attn_scores_kernel<<<(tot+255)/256,256,0,stream>>>(Q, K2, SCORES, Bn);
    attn_softmax_kernel<<<(Bn*4*12+255)/256,256,0,stream>>>(SCORES, Bn);
    size_t tv = (size_t)Bn*12*1024;
    attn_av_kernel<<<(unsigned)((tv+255)/256),256,0,stream>>>(SCORES, V, OATT, Bn); }
  gemm(stream, OATT, 1, 1024, 0, 1, out_w, 1, 1024,
       SBUF, 1, 1024, 0, 1, Bn*12, 1024, 1024, out_b, nullptr, LAT, 1024);
  layernorm_kernel<<<Bn*12, 256, 0, stream>>>(SBUF, ln_g, ln_b, LATN, 1024);
  gemm(stream, LATN, 1, 1024, 0, 1, ff1_w, 1, 1024,
       FFH, 1, 4096, 0, 1, Bn*12, 4096, 1024, ff1_b, nullptr, nullptr, 0, 2);
  gemm(stream, FFH, 1, 4096, 0, 1, ff2_w, 1, 4096,
       CAOUT, 1, 1024, 0, 1, Bn*12, 1024, 4096, ff2_b, nullptr, LATN, 1024);
  { size_t tot = (size_t)Bn*12*1024;   // [B,12,1024] -> [B,1024,12]
    transpose_b_kernel<<<(unsigned)((tot+255)/256),256,0,stream>>>(
        CAOUT, ACT0, Bn, 12, 1024); }
  // 11) gcn4 (N=B*12, c=1024, out=512) -> ACT1 = [B,512,12]
  run_gat(stream, ACT0, Bn*12, 1024, 512, g4, EDGE, E, ACT1,
          GATH, GACC, ASRC, ADST, MKEY, DEN);
  // 12) bilstm4 -> [B,1024,12]
  run_bilstm(stream, ACT1, Bn, 512, 12, 512, l4, ACT0, PRE, GATES, HST, CST);
  // 13) up1 -> [B,512,24]
  { size_t tot = (size_t)Bn*512*24;
    convt_k2s2_kernel<<<(unsigned)((tot+255)/256),256,0,stream>>>(
        ACT0, u1_w, u1_b, ACT1, Bn, 1024, 12, 512); }
  // 14) gcn5 (N=B*24, c=512, out=256) -> ACT0 = [B,256,24]
  run_gat(stream, ACT1, Bn*24, 512, 256, g5, EDGE, E, ACT0,
          GATH, GACC, ASRC, ADST, MKEY, DEN);
  // 15) bilstm5 -> [B,512,24]
  run_bilstm(stream, ACT0, Bn, 256, 24, 256, l5, ACT1, PRE, GATES, HST, CST);
  // 16) up2 -> [B,256,48]
  { size_t tot = (size_t)Bn*256*48;
    convt_k2s2_kernel<<<(unsigned)((tot+255)/256),256,0,stream>>>(
        ACT1, u2_w, u2_b, ACT0, Bn, 512, 24, 256); }
  // 17) gcn6 (N=B*48, c=256, out=128) -> ACT1 = [B,128,48]
  run_gat(stream, ACT0, Bn*48, 256, 128, g6, EDGE, E, ACT1,
          GATH, GACC, ASRC, ADST, MKEY, DEN);
  // 18) bilstm6 (H=90) -> [B,180,48]
  run_bilstm(stream, ACT1, Bn, 128, 48, 90, l6, ACT0, PRE, GATES, HST, CST);
  // 19) final: [B,180,48] -> d_out [B,90,48]
  gemm(stream, ACT0, 48, 180*48, 1, 48, fin_w, 1, 180,
       (float*)d_out, 48, 90*48, 1, 48, Bn*48, 90, 180, fin_b);
}
